// GovernanceAwareAttention_63050119905537
// MI455X (gfx1250) — compile-verified, matrix operand order checked by reading, not run on hardware
//
#include <hip/hip_runtime.h>
#include <hip/hip_bf16.h>

typedef __attribute__((ext_vector_type(16))) __bf16 v16bf;
typedef __attribute__((ext_vector_type(8)))  __bf16 v8bf;
typedef __attribute__((ext_vector_type(8)))  float  v8f;
typedef __attribute__((ext_vector_type(4)))  unsigned u32x4;
typedef __attribute__((ext_vector_type(8)))  int      i32x8;
typedef __attribute__((ext_vector_type(4)))  int      i32x4;

#define NHEADS 16
#define HD 64
#define SEQ 2048
#define BATCH 2
#define HID 1024
#define NROWS_TOTAL (BATCH * NHEADS * SEQ)   // 65536 softmax rows

__device__ __forceinline__ v8f wmma_bf16(const v16bf& a, const v16bf& b, const v8f& c) {
    return __builtin_amdgcn_wmma_f32_16x16x32_bf16(false, a, false, b, (short)0, c, false, false);
}

// Fragment layout (16-bit A/B, 16x32): lane-half selects K base {0,8}; per lane the
// 16 elements are memory-contiguous as two 8-element chunks at +0 and +16.
__device__ __forceinline__ v16bf load_frag(const __bf16* row, int laneHi) {
    const __bf16* p = row + laneHi * 8;
    v8bf lo = *(const v8bf*)(p);
    v8bf hi = *(const v8bf*)(p + 16);
    return __builtin_shufflevector(lo, hi, 0,1,2,3,4,5,6,7,8,9,10,11,12,13,14,15);
}

// ---------------------------------------------------------------------------
// TDM: async 2D tile load (global -> LDS), 2-byte elements, optional LDS row pad.
// D# packing per CDNA5 ISA §8.3/8.4. Groups 2/3 zero (tensor is 2D).
// This toolchain exposes the 6-arg builtin: (g0 u32x4, g1 i32x8, g2 i32x4,
// g3 i32x4, extra i32x8, cpol i32).
// ---------------------------------------------------------------------------
__device__ __forceinline__ void tdm_load_tile_2d_b16(
    unsigned ldsAddr, const void* gaddr,
    unsigned tensorDim0, unsigned tensorDim1, unsigned tensorStride0,
    unsigned tileDim0, unsigned tileDim1,
    unsigned padInterval, unsigned padAmount, bool padEnable)
{
    unsigned long long ga = (unsigned long long)(uintptr_t)gaddr;
    u32x4 g0;
    g0[0] = 1u;                                                   // count=1 (valid user D#)
    g0[1] = ldsAddr;                                              // LDS byte address
    g0[2] = (unsigned)ga;                                         // global_addr[31:0]
    g0[3] = (unsigned)((ga >> 32) & 0x01FFFFFFu) | (2u << 30);    // global_addr[56:32], type=2
    i32x8 g1;
    g1[0] = (int)((1u << 16) |                                    // data_size = 1 -> 2 bytes
                  (padEnable ? (1u << 20) : 0u) |                 // pad_enable
                  ((padInterval & 7u) << 22) |                    // pad_interval code
                  ((padAmount & 127u) << 25));                    // pad_amount code
    g1[1] = (int)((tensorDim0 & 0xFFFFu) << 16);                  // tensor_dim0[15:0]
    g1[2] = (int)((tensorDim0 >> 16) | ((tensorDim1 & 0xFFFFu) << 16));
    g1[3] = (int)((tensorDim1 >> 16) | ((tileDim0 & 0xFFFFu) << 16));
    g1[4] = (int)(tileDim1 & 0xFFFFu);                            // tile_dim1, tile_dim2=0
    g1[5] = (int)tensorStride0;                                   // tensor_dim0_stride[31:0]
    g1[6] = 0;                                                    // stride[47:32], dim1_stride lo
    g1[7] = 0;
    i32x4 z4 = {0, 0, 0, 0};
    i32x8 z8 = {0, 0, 0, 0, 0, 0, 0, 0};
    __builtin_amdgcn_tensor_load_to_lds(g0, g1, z4, z4, z8, 0);
}

// ---------------------------------------------------------------------------
// 0. zero the scalar output slots (entropy, influence, concentration)
// ---------------------------------------------------------------------------
__global__ void init_scalars(float* scal) {
    if (threadIdx.x < 3) scal[threadIdx.x] = 0.0f;
}

// ---------------------------------------------------------------------------
// 1. Projection GEMM:  out[b,nh,s,hd] = bf16( (X[m,:] @ W[:,n] + bias[n]) * outScale )
//    blockIdx.z selects {Q,K,V,Gk}; Q gets outScale = 1/sqrt(HD).
// ---------------------------------------------------------------------------
__global__ __launch_bounds__(128) void proj_kernel(
    const float* __restrict__ X,
    const float* W0, const float* W1, const float* W2, const float* W3,
    const float* b0, const float* b1, const float* b2, const float* b3,
    __bf16* O0, __bf16* O1, __bf16* O2, __bf16* O3)
{
    const float* W;
    const float* bias;
    __bf16* O;
    float outScale = 1.0f;
    switch (blockIdx.z) {
        case 0:  W = W0; bias = b0; O = O0; outScale = 0.125f; break;  // Q pre-scaled
        case 1:  W = W1; bias = b1; O = O1; break;
        case 2:  W = W2; bias = b2; O = O2; break;
        default: W = W3; bias = b3; O = O3; break;
    }

    __shared__ __align__(16) __bf16 Wt[64][40];   // W tile transposed: [n][k], K-contiguous

    const int tid    = threadIdx.x;
    const int wave   = tid >> 5;
    const int lane   = tid & 31;
    const int laneHi = lane >> 4;
    const int ln     = lane & 15;

    const int m0 = blockIdx.x * 64;
    const int n0 = blockIdx.y * 64;
    const int wrow = m0 + wave * 16;

    v8f acc[4];
#pragma unroll
    for (int nt = 0; nt < 4; ++nt) acc[nt] = (v8f){0.f,0.f,0.f,0.f,0.f,0.f,0.f,0.f};

    const int sr = (tid >> 3) & 15;   // staging row (k) within 16
    const int sc = (tid & 7) * 8;     // staging col (n) chunk

    for (int k0 = 0; k0 < HID; k0 += 32) {
        __syncthreads();
        // stage W tile 32x64 f32 -> bf16, transposed into Wt[n][k]
#pragma unroll
        for (int rr = 0; rr < 32; rr += 16) {
            const float* wp = W + (size_t)(k0 + rr + sr) * HID + n0 + sc;
            float4 w01 = *(const float4*)(wp);
            float4 w23 = *(const float4*)(wp + 4);
            Wt[sc + 0][rr + sr] = (__bf16)w01.x;
            Wt[sc + 1][rr + sr] = (__bf16)w01.y;
            Wt[sc + 2][rr + sr] = (__bf16)w01.z;
            Wt[sc + 3][rr + sr] = (__bf16)w01.w;
            Wt[sc + 4][rr + sr] = (__bf16)w23.x;
            Wt[sc + 5][rr + sr] = (__bf16)w23.y;
            Wt[sc + 6][rr + sr] = (__bf16)w23.z;
            Wt[sc + 7][rr + sr] = (__bf16)w23.w;
        }
        __syncthreads();

        // A fragment straight from global: 4 x float4, packed to bf16
        const float* xp = X + (size_t)(wrow + ln) * HID + k0 + laneHi * 8;
        v16bf a;
        {
            float4 f0 = *(const float4*)(xp);
            float4 f1 = *(const float4*)(xp + 4);
            float4 f2 = *(const float4*)(xp + 16);
            float4 f3 = *(const float4*)(xp + 20);
            a[0]  = (__bf16)f0.x;  a[1]  = (__bf16)f0.y;
            a[2]  = (__bf16)f0.z;  a[3]  = (__bf16)f0.w;
            a[4]  = (__bf16)f1.x;  a[5]  = (__bf16)f1.y;
            a[6]  = (__bf16)f1.z;  a[7]  = (__bf16)f1.w;
            a[8]  = (__bf16)f2.x;  a[9]  = (__bf16)f2.y;
            a[10] = (__bf16)f2.z;  a[11] = (__bf16)f2.w;
            a[12] = (__bf16)f3.x;  a[13] = (__bf16)f3.y;
            a[14] = (__bf16)f3.z;  a[15] = (__bf16)f3.w;
        }

#pragma unroll
        for (int nt = 0; nt < 4; ++nt) {
            v16bf bb = load_frag(&Wt[nt * 16 + ln][0], laneHi);
            acc[nt] = wmma_bf16(a, bb, acc[nt]);
        }
    }

    // epilogue: bias add + scale, write bf16 in [B, NH, S, HD] layout
#pragma unroll
    for (int nt = 0; nt < 4; ++nt) {
#pragma unroll
        for (int r = 0; r < 8; ++r) {
            int m = wrow + laneHi * 8 + r;          // 0..4095
            int c = n0 + nt * 16 + ln;              // 0..1023
            float val = (acc[nt][r] + bias[c]) * outScale;
            int b  = m >> 11, s = m & 2047;
            int nh = c >> 6,  hd = c & 63;
            O[(((size_t)(b * NHEADS + nh)) * SEQ + s) * HD + hd] = (__bf16)val;
        }
    }
}

// ---------------------------------------------------------------------------
// 2. gq = governance_embeddings @ Wgq + bgq   ([2,1024] f32)
// ---------------------------------------------------------------------------
__global__ __launch_bounds__(256) void gq_kernel(
    const float* __restrict__ ge, const float* __restrict__ Wgq,
    const float* __restrict__ bgq, float* __restrict__ gq)
{
    int idx = blockIdx.x * 256 + threadIdx.x;   // 0..2047
    int b = idx >> 10, n = idx & 1023;
    float acc = bgq[n];
    for (int k = 0; k < HID; ++k)
        acc = fmaf(ge[b * HID + k], Wgq[(size_t)k * HID + n], acc);
    gq[idx] = acc;
}

// ---------------------------------------------------------------------------
// 3. influence = mean |gq[b,h,:] . gk[b,s,h,:]| over (b,h,s)
// ---------------------------------------------------------------------------
__global__ __launch_bounds__(256) void influence_kernel(
    const float* __restrict__ gq, const __bf16* __restrict__ Gk,
    float* __restrict__ infl)
{
    int idx = blockIdx.x * 256 + threadIdx.x;   // (b*16+h)*2048 + s
    int h = (idx >> 11) & 15;
    int b = idx >> 15;
    const __bf16* gk = Gk + (size_t)idx * HD;
    const float*  q  = gq + b * HID + h * HD;
    float acc = 0.f;
#pragma unroll
    for (int d = 0; d < HD; d += 8) {
        v8bf g = *(const v8bf*)(gk + d);
#pragma unroll
        for (int j = 0; j < 8; ++j) acc = fmaf(q[d + j], (float)g[j], acc);
    }
    acc = fabsf(acc) * (1.0f / (float)NROWS_TOTAL);

    // block reduce
#pragma unroll
    for (int off = 16; off >= 1; off >>= 1) acc += __shfl_xor(acc, off, 32);
    __shared__ float red[8];
    if ((threadIdx.x & 31) == 0) red[threadIdx.x >> 5] = acc;
    __syncthreads();
    if (threadIdx.x == 0) {
        float v = 0.f;
#pragma unroll
        for (int i = 0; i < 8; ++i) v += red[i];
        atomicAdd(infl, v);
    }
}

// ---------------------------------------------------------------------------
// 4. Flash attention per (b, h, 64-query tile). 128 threads = 4 waves.
//    K tile staged by the Tensor Data Mover (async DMA + LDS row padding);
//    V tile staged transposed by VALU (TDM cannot transpose).
//    Q is pre-scaled by 1/sqrt(HD); gov bias is softmax-invariant -> dropped.
//    entropy_row = m + log(l) - t/l ; concentration_row = 1/l.
// ---------------------------------------------------------------------------
__global__ __launch_bounds__(128) void attn_kernel(
    const __bf16* __restrict__ Qb, const __bf16* __restrict__ Kb,
    const __bf16* __restrict__ Vb, float* __restrict__ ctx,
    float* __restrict__ ent_out, float* __restrict__ conc_out)
{
    const int bh = blockIdx.y;              // b*16 + h
    const int b  = bh >> 4, h = bh & 15;
    const int q0 = blockIdx.x * 64;

    const int tid    = threadIdx.x;
    const int wave   = tid >> 5;
    const int lane   = tid & 31;
    const int laneHi = lane >> 4;
    const int ln     = lane & 15;

    const __bf16* Qh = Qb + (size_t)bh * SEQ * HD;
    const __bf16* Kh = Kb + (size_t)bh * SEQ * HD;
    const __bf16* Vh = Vb + (size_t)bh * SEQ * HD;

    __shared__ __align__(16) __bf16 Kl[32][72];        // [key][d]; TDM pads 64+8 per row
    __shared__ __align__(16) __bf16 Vt[64][40];        // [d][key]  (K-contiguous for PV B-frag)
    __shared__ __align__(16) __bf16 Pl[4][16][40];     // per-wave P bounce, [row][key]

    const unsigned klBase = (unsigned)(uintptr_t)(&Kl[0][0]);  // LDS byte offset (addr[31:0])

    // Q fragments (16 rows x 64 K) live in registers for the whole key loop
    const __bf16* qp = Qh + (size_t)(q0 + wave * 16 + ln) * HD;
    v16bf qa[2];
    qa[0] = load_frag(qp,      laneHi);
    qa[1] = load_frag(qp + 32, laneHi);

    v8f o[4];
#pragma unroll
    for (int nt = 0; nt < 4; ++nt) o[nt] = (v8f){0.f,0.f,0.f,0.f,0.f,0.f,0.f,0.f};
    float mrow[8], lrow[8], trow[8];
#pragma unroll
    for (int r = 0; r < 8; ++r) { mrow[r] = -1.0e30f; lrow[r] = 0.f; trow[r] = 0.f; }

    for (int j0 = 0; j0 < SEQ; j0 += 32) {
        __syncthreads();
        // K tile 32x64 bf16 via Tensor Data Mover (one issue per block; EXEC-ignored).
        // pad_interval code 4 -> 32 DWORDs (one 128B row), pad_amount code 3 -> 4 DWORDs (16B),
        // reproducing the Kl[32][72] padded layout.
        if (wave == 0) {
            tdm_load_tile_2d_b16(klBase, Kh + (size_t)j0 * HD,
                                 /*tensorDim0=*/HD, /*tensorDim1=*/SEQ, /*stride0=*/HD,
                                 /*tileDim0=*/HD, /*tileDim1=*/32,
                                 /*padInterval=*/4, /*padAmount=*/3, /*padEnable=*/true);
        }
        // V tile staged transposed: 16B vector global loads, scatter into Vt[d][key]
#pragma unroll
        for (int p = 0; p < 2; ++p) {
            int e = (tid + p * 128) * 8;     // element index in 32x64 tile
            int r = e >> 6, c = e & 63;
            v8bf vv = *(const v8bf*)(Vh + (size_t)(j0 + r) * HD + c);
#pragma unroll
            for (int k = 0; k < 8; ++k) Vt[c + k][r] = vv[k];
        }
        if (j0 + 32 < SEQ) {
            __builtin_prefetch(Vh + (size_t)(j0 + 32 + (tid >> 2)) * HD, 0, 1);
        }
        if (wave == 0) __builtin_amdgcn_s_wait_tensorcnt(0);
        __syncthreads();

        // scores: 16 rows x 32 keys = two 16x16 C-frags, K-dim 64 = 2 chunks
        v8f sf[2];
        sf[0] = (v8f){0.f,0.f,0.f,0.f,0.f,0.f,0.f,0.f};
        sf[1] = sf[0];
#pragma unroll
        for (int nt = 0; nt < 2; ++nt)
#pragma unroll
            for (int kc = 0; kc < 2; ++kc) {
                v16bf bb = load_frag(&Kl[nt * 16 + ln][kc * 32], laneHi);  // B[kk][n]=K[key n][kk]
                sf[nt] = wmma_bf16(qa[kc], bb, sf[nt]);
            }

        // online softmax (rows replicated across the 16 lanes of each half)
#pragma unroll
        for (int r = 0; r < 8; ++r) {
            float v0 = sf[0][r];
            float v1 = sf[1][r];
            float mx = fmaxf(v0, v1);
#pragma unroll
            for (int off = 1; off < 16; off <<= 1) mx = fmaxf(mx, __shfl_xor(mx, off, 32));
            float mnew  = fmaxf(mrow[r], mx);
            float alpha = __expf(mrow[r] - mnew);
            float p0 = __expf(v0 - mnew);
            float p1 = __expf(v1 - mnew);
            float ps = p0 + p1;
            float ts = fmaf(p0, v0, p1 * v1);
#pragma unroll
            for (int off = 1; off < 16; off <<= 1) {
                ps += __shfl_xor(ps, off, 32);
                ts += __shfl_xor(ts, off, 32);
            }
            lrow[r] = lrow[r] * alpha + ps;
            trow[r] = trow[r] * alpha + ts;
            mrow[r] = mnew;
#pragma unroll
            for (int nt = 0; nt < 4; ++nt) o[nt][r] *= alpha;
            // bounce P through LDS (C-frag layout -> A-frag layout)
            Pl[wave][laneHi * 8 + r][ln]      = (__bf16)p0;
            Pl[wave][laneHi * 8 + r][16 + ln] = (__bf16)p1;
        }

        // A-frag of P (16x32), then O += P @ V
        v16bf pa = load_frag(&Pl[wave][ln][0], laneHi);
#pragma unroll
        for (int nt = 0; nt < 4; ++nt) {
            v16bf bb = load_frag(&Vt[nt * 16 + ln][0], laneHi);  // B[kk][n]=Vt[d n][key kk]
            o[nt] = wmma_bf16(pa, bb, o[nt]);
        }
    }

    // epilogue: normalize, write context [b,s,h*64+d] f32 + scalar stats
    const float inv_rows = 1.0f / (float)NROWS_TOTAL;
#pragma unroll
    for (int r = 0; r < 8; ++r) {
        int qm = q0 + wave * 16 + laneHi * 8 + r;
        float linv = 1.0f / lrow[r];
#pragma unroll
        for (int nt = 0; nt < 4; ++nt) {
            ctx[((size_t)b * SEQ + qm) * HID + h * HD + nt * 16 + ln] = o[nt][r] * linv;
        }
        if (ln == 0) {
            float ent = mrow[r] + __logf(lrow[r]) - trow[r] * linv;
            atomicAdd(ent_out,  ent  * inv_rows);
            atomicAdd(conc_out, linv * inv_rows);   // max prob = exp(m-m)/l = 1/l
        }
    }
}

// ---------------------------------------------------------------------------
// launch
// ---------------------------------------------------------------------------
extern "C" void kernel_launch(void* const* d_in, const int* in_sizes, int n_in,
                              void* d_out, int out_size, void* d_ws, size_t ws_size,
                              hipStream_t stream) {
    const float* hs  = (const float*)d_in[0];
    const float* ge  = (const float*)d_in[1];
    const float* Wq  = (const float*)d_in[2];
    const float* bq  = (const float*)d_in[3];
    const float* Wk  = (const float*)d_in[4];
    const float* bk  = (const float*)d_in[5];
    const float* Wv  = (const float*)d_in[6];
    const float* bv  = (const float*)d_in[7];
    const float* Wgq = (const float*)d_in[8];
    const float* bgq = (const float*)d_in[9];
    const float* Wgk = (const float*)d_in[10];
    const float* bgk = (const float*)d_in[11];

    float* out  = (float*)d_out;
    float* ctx  = out;                               // [B,S,H]
    float* scal = out + (size_t)BATCH * SEQ * HID;   // entropy, influence, concentration

    const size_t proj_elems = (size_t)BATCH * NHEADS * SEQ * HD;   // 4M
    char* ws = (char*)d_ws;
    __bf16* Qb  = (__bf16*)(ws);
    __bf16* Kb  = (__bf16*)(ws + proj_elems * 2);
    __bf16* Vb  = (__bf16*)(ws + proj_elems * 4);
    __bf16* Gkb = (__bf16*)(ws + proj_elems * 6);
    float*  gq  = (float*)(ws + proj_elems * 8);                   // [2,1024]

    init_scalars<<<1, 32, 0, stream>>>(scal);

    proj_kernel<<<dim3(64, 16, 4), 128, 0, stream>>>(
        hs, Wq, Wk, Wv, Wgk, bq, bk, bv, bgk, Qb, Kb, Vb, Gkb);

    gq_kernel<<<8, 256, 0, stream>>>(ge, Wgq, bgq, gq);

    influence_kernel<<<NROWS_TOTAL / 256, 256, 0, stream>>>(gq, Gkb, scal + 1);

    attn_kernel<<<dim3(SEQ / 64, BATCH * NHEADS), 128, 0, stream>>>(
        Qb, Kb, Vb, ctx, scal + 0, scal + 2);
}
